// MoELayerOptimized_3719441678904
// MI455X (gfx1250) — matrix operand order repaired and checked by reading, not compiled, not run on hardware
//
#include <hip/hip_runtime.h>

// MoE top-2 (B=1, S=2048, H=1024, I=4096, E=8), fp32 in/out.
//   out[t] = sum_{k<2} wt[t,k] * ( silu(x[t] @ w1[e_k]^T) @ w2[e_k]^T )
// Compute on v_wmma_f32_16x16x32_bf16; fp32 weights converted on the fly with
// packed v_cvt_pk_bf16_f32 (via __builtin_convertvector), b128 global/LDS
// fragment loads. Per-expert weights (32MB) are L2-resident across tiles.

#define TT 2048
#define HH 1024
#define II 4096
#define EE 8

typedef __attribute__((ext_vector_type(16))) __bf16 v16bf;
typedef __attribute__((ext_vector_type(2)))  __bf16 v2bf;
typedef __attribute__((ext_vector_type(8)))  float  v8f;
typedef __attribute__((ext_vector_type(2)))  float  v2f;
typedef __attribute__((ext_vector_type(8)))  unsigned v8u;

// ---------- helpers ----------

// 2x fp32 -> packed bf16 (RNE) : lowers to v_cvt_pk_bf16_f32
__device__ __forceinline__ unsigned pack2_bf16(float a, float b) {
    v2f f = {a, b};
    v2bf h = __builtin_convertvector(f, v2bf);
    return __builtin_bit_cast(unsigned, h);
}
__device__ __forceinline__ unsigned short bf16_1(float a) {
    return (unsigned short)(pack2_bf16(a, a) & 0xFFFFu);
}

// A-operand (16x32 bf16) from LDS packed-pair array (16B-aligned rows).
// Contiguous layout: lanes 0-15: M=lane, K=0..15;
// lanes 16-31: M=lane-16, K=16..31. Pair p holds K=2p,2p+1.
__device__ __forceinline__ v16bf load_a_frag(const unsigned* row0,
                                             int rowStrideU, int lane) {
    const unsigned* row =
        row0 + (lane & 15) * rowStrideU + ((lane < 16) ? 0 : 8);
    uint4 lo = *(const uint4*)row;          // pairs pb..pb+3
    uint4 hi = *(const uint4*)(row + 4);    // pairs pb+4..pb+7
    v8u r = {lo.x, lo.y, lo.z, lo.w, hi.x, hi.y, hi.z, hi.w};
    return __builtin_bit_cast(v16bf, r);
}

// B-operand (32x16 bf16) built per-lane from a global fp32 row.
// V0..7: lanes0-15 K=0..15, lanes16-31 K=16..31 (caller adds kbase).
// The 16 floats are contiguous & 64B-aligned -> 4x global_load_b128.
__device__ __forceinline__ v16bf load_b_frag_g(const float* __restrict__ p) {
    float4 q0 = *(const float4*)(p);
    float4 q1 = *(const float4*)(p + 4);
    float4 q2 = *(const float4*)(p + 8);
    float4 q3 = *(const float4*)(p + 12);
    v8u r = {pack2_bf16(q0.x, q0.y), pack2_bf16(q0.z, q0.w),
             pack2_bf16(q1.x, q1.y), pack2_bf16(q1.z, q1.w),
             pack2_bf16(q2.x, q2.y), pack2_bf16(q2.z, q2.w),
             pack2_bf16(q3.x, q3.y), pack2_bf16(q3.z, q3.w)};
    return __builtin_bit_cast(v16bf, r);
}

__device__ __forceinline__ v8f wmma_bf16(v16bf a, v16bf b, v8f c) {
    return __builtin_amdgcn_wmma_f32_16x16x32_bf16(
        false, a, false, b, (short)0, c, false, false);
}

// ---------- kernel 1: gating (one wave32 per token) ----------

__global__ __launch_bounds__(256) void moe_gate(
    const float* __restrict__ x, const float* __restrict__ gate_w,
    int* __restrict__ sel_e, float* __restrict__ sel_w) {
    int wave = threadIdx.x >> 5;
    int lane = threadIdx.x & 31;
    int t = blockIdx.x * 8 + wave;           // grid = 256 -> 2048 tokens

    float acc[EE];
#pragma unroll
    for (int e = 0; e < EE; ++e) acc[e] = 0.f;

    const float* xr = x + (size_t)t * HH;
    for (int h = lane; h < HH; h += 32) {
        float xv = xr[h];
#pragma unroll
        for (int e = 0; e < EE; ++e) acc[e] += xv * gate_w[e * HH + h];
    }
#pragma unroll
    for (int e = 0; e < EE; ++e)
        for (int o = 16; o > 0; o >>= 1)
            acc[e] += __shfl_xor(acc[e], o, 32);

    if (lane == 0) {
        int e0 = 0;
#pragma unroll
        for (int e = 1; e < EE; ++e) if (acc[e] > acc[e0]) e0 = e;
        int e1 = (e0 == 0) ? 1 : 0;
#pragma unroll
        for (int e = 0; e < EE; ++e)
            if (e != e0 && acc[e] > acc[e1]) e1 = e;
        float m  = acc[e0];
        float p0 = __expf(acc[e0] - m);
        float p1 = __expf(acc[e1] - m);
        float inv = 1.f / (p0 + p1);
        sel_e[2 * t]     = e0;  sel_w[2 * t]     = p0 * inv;
        sel_e[2 * t + 1] = e1;  sel_w[2 * t + 1] = p1 * inv;
    }
}

// ---------- kernel 2: deterministic per-expert compaction ----------

__global__ void moe_build_lists(
    const int* __restrict__ sel_e, const float* __restrict__ sel_w,
    int* __restrict__ counts, int* __restrict__ tok_list,
    float* __restrict__ wt_list) {
    int e = threadIdx.x;
    if (e >= EE) return;
    int cnt = 0;
    for (int t = 0; t < TT; ++t) {
#pragma unroll
        for (int k = 0; k < 2; ++k) {
            if (sel_e[2 * t + k] == e) {
                tok_list[e * TT + cnt] = 2 * t + k;   // token*2 | slot
                wt_list[e * TT + cnt]  = sel_w[2 * t + k];
                ++cnt;
            }
        }
    }
    counts[e] = cnt;
}

// ---------- kernel 3: expert FFN on WMMA bf16 ----------
// grid = E * 128 tiles, 256 threads = 8 waves.
// Wave w owns H columns [128w,128w+128); hid chunk rows [32w,32w+32).

#define XSTRIDE  516   // uints per x_pairs row (16B aligned, bank-rotating)
#define HSTRIDE  132   // uints per hid row     (16B aligned, bank-rotating)

__global__ __launch_bounds__(256) void moe_ffn(
    const float* __restrict__ x,
    const float* __restrict__ w1,   // [E][I][H]
    const float* __restrict__ w2,   // [E][H][I]
    const int* __restrict__ counts,
    const int* __restrict__ tok_list,
    const float* __restrict__ wt_list,
    float* __restrict__ partial)    // [2][T][H]
{
    int e    = blockIdx.x >> 7;
    int tile = blockIdx.x & 127;
    int cnt  = counts[e];
    int base = tile * 16;
    if (base >= cnt) return;                 // wave-uniform early out
    int nvalid = min(16, cnt - base);

    __shared__ unsigned       x_pairs[16][XSTRIDE];      // ~33 KB
    __shared__ unsigned short hid_lds[16][2 * HSTRIDE];  // ~8.4 KB
    __shared__ int   tokk[16];
    __shared__ float wt16[16];

    int tid  = threadIdx.x;
    int lane = tid & 31;
    int wave = tid >> 5;
    int col  = lane & 15;
    int kbase = (lane < 16) ? 0 : 16;
    int rbase = (lane >> 4) << 3;

    if (tid < 16) {
        int entry = 0; float w = 0.f;
        if (tid < nvalid) {
            entry = tok_list[e * TT + base + tid];
            w     = wt_list[e * TT + base + tid];
        }
        tokk[tid] = entry;
        wt16[tid] = w;
    }
    __syncthreads();

    // stage x tile into LDS as packed bf16 pairs (float4 reads)
    for (int idx = tid; idx < 16 * 256; idx += 256) {
        int m = idx >> 8;
        int p = idx & 255;
        float4 q = *(const float4*)(x + (size_t)(tokk[m] >> 1) * HH + 4 * p);
        x_pairs[m][2 * p]     = pack2_bf16(q.x, q.y);
        x_pairs[m][2 * p + 1] = pack2_bf16(q.z, q.w);
    }
    __syncthreads();

    v8f accB[8];
#pragma unroll
    for (int nt = 0; nt < 8; ++nt)
        accB[nt] = (v8f){0.f, 0.f, 0.f, 0.f, 0.f, 0.f, 0.f, 0.f};

    const unsigned* hid_u = (const unsigned*)&hid_lds[0][0];

#pragma unroll 1
    for (int ic = 0; ic < II; ic += 256) {
        // ---- stage A: hid[16 x 32] = x_tile @ w1_rows^T, K = 1024 ----
        v8f h0 = (v8f){0.f, 0.f, 0.f, 0.f, 0.f, 0.f, 0.f, 0.f};
        v8f h1 = h0;
        const float* w1r0 =
            w1 + ((size_t)e * II + ic + 32 * wave + col) * HH + kbase;
        const float* w1r1 = w1r0 + (size_t)16 * HH;
#pragma unroll 4
        for (int kc = 0; kc < HH; kc += 32) {
            v16bf af = load_a_frag(&x_pairs[0][kc >> 1], XSTRIDE, lane);
            v16bf b0 = load_b_frag_g(w1r0 + kc);
            v16bf b1 = load_b_frag_g(w1r1 + kc);
            h0 = wmma_bf16(af, b0, h0);
            h1 = wmma_bf16(af, b1, h1);
        }
        __syncthreads();   // all waves done reading hid_lds (prev iter)
#pragma unroll
        for (int v = 0; v < 8; ++v) {
            int m = rbase + v;
            float s0 = h0[v]; s0 = s0 / (1.f + __expf(-s0));
            float s1 = h1[v]; s1 = s1 / (1.f + __expf(-s1));
            hid_lds[m][32 * wave + col]      = bf16_1(s0);
            hid_lds[m][32 * wave + 16 + col] = bf16_1(s1);
        }
        __syncthreads();   // hid chunk ready

        // ---- stage B: out[16 x 128] += hid[16 x 256] @ w2_cols^T ----
#pragma unroll 2
        for (int ks = 0; ks < 256; ks += 32) {
            v16bf af = load_a_frag(hid_u + (ks >> 1), HSTRIDE, lane);
#pragma unroll
            for (int nt = 0; nt < 8; ++nt) {
                const float* w2r =
                    w2 + ((size_t)e * HH + 128 * wave + nt * 16 + col) * II
                       + ic + ks + kbase;
                v16bf bf_ = load_b_frag_g(w2r);
                accB[nt] = wmma_bf16(af, bf_, accB[nt]);
            }
        }
    }

    // ---- epilogue: scale by routing weight, scatter to partial[k][t][h] ----
#pragma unroll
    for (int nt = 0; nt < 8; ++nt) {
        int h = 128 * wave + nt * 16 + col;
#pragma unroll
        for (int v = 0; v < 8; ++v) {
            int m = rbase + v;
            if (m < nvalid) {
                int entry = tokk[m];
                int t = entry >> 1, k = entry & 1;
                partial[((size_t)k * TT + t) * HH + h] = accB[nt][v] * wt16[m];
            }
        }
    }
}

// ---------- kernel 4: combine the two expert slots ----------

__global__ void moe_final(const float* __restrict__ partial,
                          float* __restrict__ out) {
    int idx = blockIdx.x * 256 + threadIdx.x;
    if (idx < TT * HH)
        out[idx] = partial[idx] + partial[(size_t)TT * HH + idx];
}

// ---------- launch ----------
// Workspace layout (bytes):
//   sel_e @0 (16K) | sel_w @16384 (16K) | counts @32768 (32)
//   tok_list @65536 (64K) | wt_list @131072 (64K) | partial @196608 (16MB)

extern "C" void kernel_launch(void* const* d_in, const int* in_sizes, int n_in,
                              void* d_out, int out_size, void* d_ws,
                              size_t ws_size, hipStream_t stream) {
    const float* x      = (const float*)d_in[0];
    const float* gate_w = (const float*)d_in[1];
    const float* w1     = (const float*)d_in[2];
    const float* w2     = (const float*)d_in[3];
    float* out = (float*)d_out;

    char* ws = (char*)d_ws;
    int*   sel_e    = (int*)(ws + 0);
    float* sel_w    = (float*)(ws + 16384);
    int*   counts   = (int*)(ws + 32768);
    int*   tok_list = (int*)(ws + 65536);
    float* wt_list  = (float*)(ws + 131072);
    float* partial  = (float*)(ws + 196608);

    moe_gate<<<TT / 8, 256, 0, stream>>>(x, gate_w, sel_e, sel_w);
    moe_build_lists<<<1, 32, 0, stream>>>(sel_e, sel_w, counts, tok_list,
                                          wt_list);
    moe_ffn<<<EE * (TT / 16), 256, 0, stream>>>(x, w1, w2, counts, tok_list,
                                                wt_list, partial);
    moe_final<<<(TT * HH) / 256, 256, 0, stream>>>(partial, out);
}
